// LSTMCellWithNorm_69595650064874
// MI455X (gfx1250) — compile-verified, matrix operand order checked by reading, not run on hardware
//
#include <hip/hip_runtime.h>

typedef __attribute__((ext_vector_type(16))) __bf16 v16bf;
typedef __attribute__((ext_vector_type(8)))  float  v8f;

#define D_IN  512
#define HH    512
#define FANIN 1024
#define MTILE 16
#define EPSV  1e-5f

__device__ __forceinline__ float fast_sigmoid(float x) {
    return 1.0f / (1.0f + __expf(-x));
}
__device__ __forceinline__ float fast_tanh(float x) {
    float e = __expf(-2.0f * x);
    return (1.0f - e) / (1.0f + e);
}
__device__ __forceinline__ void wave_reduce2(float& s, float& s2) {
    #pragma unroll
    for (int m = 16; m >= 1; m >>= 1) {
        s  += __shfl_xor(s,  m, 32);
        s2 += __shfl_xor(s2, m, 32);
    }
}

// ---------------------------------------------------------------------------
// Pre-pass: fp32 weights [512 x 1024] (row-major, W[n][k]) -> bf16 fragments
// pre-swizzled into the WMMA 16-bit B-matrix (32x16 KxN) lane layout.
// Fragment id = gn*32 + kb, gn = gate*32 + ntile (128 gn total), kb = K/32.
// Packed element p = lane*16 + elem; lane = h*16 + n; elem = 2*v + e;
// K = (v>=4?16:0) + 8*h + 2*(v&3) + e   (per ISA 16-bit A/B VGPR table).
// ---------------------------------------------------------------------------
__global__ void pack_weights_kernel(const float* __restrict__ Wf,
                                    const float* __restrict__ Wi,
                                    const float* __restrict__ Wo,
                                    const float* __restrict__ Wj,
                                    __bf16* __restrict__ wpack) {
    int frag = blockIdx.x;            // 0..4095
    int gn   = frag >> 5;
    int kb   = frag & 31;
    int gate = gn >> 5;
    int nig  = gn & 31;
    const float* W = (gate == 0) ? Wf : (gate == 1) ? Wi : (gate == 2) ? Wo : Wj;

    int p    = threadIdx.x;           // 0..511
    int lane = p >> 4;
    int elem = p & 15;
    int h    = lane >> 4;
    int n    = lane & 15;
    int v    = elem >> 1;
    int e    = elem & 1;
    int K    = ((v >= 4) ? 16 : 0) + h * 8 + ((v & 3) << 1) + e;

    wpack[(size_t)frag * 512 + p] = (__bf16)W[(nig * 16 + n) * FANIN + kb * 32 + K];
}

// ---------------------------------------------------------------------------
// Fused LSTM+LayerNorm kernel. 512 threads = 16 wave32s per 16-row tile.
// Wave w owns gate columns [w*128, w*128+128) (8 WMMA N-tiles), gate-major.
// K-loop is software-pipelined with sched_barrier fences so next-step loads
// are issued ahead of the current WMMA chain (partial loadcnt waits).
// ---------------------------------------------------------------------------
extern __shared__ char smem[];

__global__ void __launch_bounds__(512, 1)
lstm_kernel(const float* __restrict__ x,  const float* __restrict__ ph,
            const float* __restrict__ pc, const __bf16* __restrict__ wpack,
            const float* __restrict__ bf_, const float* __restrict__ bi_,
            const float* __restrict__ bo_, const float* __restrict__ bj_,
            const float* __restrict__ gf,  const float* __restrict__ btf,
            const float* __restrict__ gi,  const float* __restrict__ bti,
            const float* __restrict__ go,  const float* __restrict__ bto,
            const float* __restrict__ gj,  const float* __restrict__ btj,
            const float* __restrict__ gc,  const float* __restrict__ btc,
            float* __restrict__ out_h, float* __restrict__ out_c) {
    __bf16* a_s  = (__bf16*)smem;                 // 32 KiB: A fragments
    float*  gbuf = (float*)(smem + 32768);        // 128 KiB: 16 x 2048 gate vals
    float*  cbuf = (float*)smem;                  // reuse A region after GEMM

    const int tid  = threadIdx.x;
    const int lane = tid & 31;
    const int wave = tid >> 5;
    const int m0   = blockIdx.x * MTILE;

    // ---- Stage A: combined=[x,prev_h] -> bf16 in LDS, A-fragment swizzle ----
    #pragma unroll 4
    for (int i = 0; i < 32; ++i) {
        int q = i * 512 + tid;        // 0..16383
        int m = q >> 10;              // row 0..15
        int k = q & 1023;             // K 0..1023
        float val = (k < D_IN) ? x[(size_t)(m0 + m) * D_IN + k]
                               : ph[(size_t)(m0 + m) * HH + (k - D_IN)];
        int kb  = k >> 5;
        int kin = k & 31;
        int grp = kin >> 3;
        int hh  = grp & 1;
        int vhi = grp >> 1;
        int r   = kin & 7;
        int vv  = vhi * 4 + (r >> 1);
        int ee  = r & 1;
        int lane_a = hh * 16 + m;
        a_s[kb * 512 + lane_a * 16 + 2 * vv + ee] = (__bf16)val;
    }
    __syncthreads();

    // ---- GEMM: 8 accumulator tiles per wave, software-pipelined K loop ----
    v8f acc[8];
    const v8f vzero = {0.f, 0.f, 0.f, 0.f, 0.f, 0.f, 0.f, 0.f};
    #pragma unroll
    for (int j = 0; j < 8; ++j) acc[j] = vzero;

    const __bf16* wbase = wpack + (size_t)(wave * 8) * 32 * 512;
    const __bf16* abase = a_s + lane * 16;
    const int     boff  = lane * 16;

    v16bf aA, aB, bA[8], bB[8];
    aA = *(const v16bf*)(abase + 0 * 512);
    #pragma unroll
    for (int j = 0; j < 8; ++j)
        bA[j] = *(const v16bf*)(wbase + (size_t)(j * 32 + 0) * 512 + boff);

    for (int kb = 0; kb < 32; kb += 2) {
        // -- prefetch step kb+1 --
        aB = *(const v16bf*)(abase + (kb + 1) * 512);
        #pragma unroll
        for (int j = 0; j < 8; ++j)
            bB[j] = *(const v16bf*)(wbase + (size_t)(j * 32 + (kb + 1)) * 512 + boff);
        __builtin_amdgcn_sched_barrier(0);

        // -- consume step kb --
        #pragma unroll
        for (int j = 0; j < 8; ++j)
            acc[j] = __builtin_amdgcn_wmma_f32_16x16x32_bf16(
                false, aA, false, bA[j], (short)0, acc[j], false, false);
        __builtin_amdgcn_sched_barrier(0);

        // -- prefetch step kb+2 (wraps to 0 on final iteration; harmless) --
        {
            int kn = (kb + 2) & 31;
            aA = *(const v16bf*)(abase + kn * 512);
            #pragma unroll
            for (int j = 0; j < 8; ++j)
                bA[j] = *(const v16bf*)(wbase + (size_t)(j * 32 + kn) * 512 + boff);
        }
        __builtin_amdgcn_sched_barrier(0);

        // -- consume step kb+1 --
        #pragma unroll
        for (int j = 0; j < 8; ++j)
            acc[j] = __builtin_amdgcn_wmma_f32_16x16x32_bf16(
                false, aB, false, bB[j], (short)0, acc[j], false, false);
        __builtin_amdgcn_sched_barrier(0);
    }

    // ---- Store accumulators + bias into gate buffer ----
    const float* biasArr[4] = {bf_, bi_, bo_, bj_};
    #pragma unroll
    for (int j = 0; j < 8; ++j) {
        int gn   = wave * 8 + j;
        int gate = gn >> 5;
        int gcol = gn * 16 + (lane & 15);           // 0..2047, gate-major
        float bv = biasArr[gate][gcol - gate * 512];
        int mh   = (lane >> 4) * 8;                 // C/D layout: M = r + 8*(lane/16)
        #pragma unroll
        for (int r = 0; r < 8; ++r)
            gbuf[(mh + r) * 2048 + gcol] = acc[j][r] + bv;
    }
    __syncthreads();

    // ---- Gate LayerNorm + activation (one wave per row-gate) ----
    const float* gamArr[4] = {gf, gi, go, gj};
    const float* betArr[4] = {btf, bti, bto, btj};
    #pragma unroll 1
    for (int g = 0; g < 4; ++g) {
        int m = wave;
        int base = m * 2048 + g * 512;
        float v[16], s = 0.f, s2 = 0.f;
        #pragma unroll
        for (int i = 0; i < 16; ++i) {
            v[i] = gbuf[base + i * 32 + lane];
            s += v[i];
            s2 += v[i] * v[i];
        }
        wave_reduce2(s, s2);
        float mean = s * (1.0f / 512.0f);
        float var  = fmaxf((s2 - 512.0f * mean * mean) * (1.0f / 511.0f), 0.0f);
        float rden = 1.0f / (sqrtf(var) + EPSV);    // eps added to std (ddof=1)
        const float* gamma = gamArr[g];
        const float* beta  = betArr[g];
        #pragma unroll
        for (int i = 0; i < 16; ++i) {
            int col = i * 32 + lane;
            float y = gamma[col] * ((v[i] - mean) * rden) + beta[col];
            y = (g == 3) ? fast_tanh(y) : fast_sigmoid(y);
            gbuf[base + col] = y;
        }
    }
    __syncthreads();

    // ---- c_raw = ft*prev_c + min(1-ft, it)*jt ----
    #pragma unroll 1
    for (int m = 0; m < 16; ++m) {
        int col = tid;
        float ft = gbuf[m * 2048 + col];
        float ig = gbuf[m * 2048 + 512 + col];
        float jt = gbuf[m * 2048 + 1536 + col];
        float cp = pc[(size_t)(m0 + m) * HH + col];
        cbuf[m * 512 + col] = ft * cp + fminf(1.0f - ft, ig) * jt;
    }
    __syncthreads();

    // ---- LayerNorm(c), h = ot * c ; write outputs ----
    {
        int m = wave;
        float v[16], s = 0.f, s2 = 0.f;
        #pragma unroll
        for (int i = 0; i < 16; ++i) {
            v[i] = cbuf[m * 512 + i * 32 + lane];
            s += v[i];
            s2 += v[i] * v[i];
        }
        wave_reduce2(s, s2);
        float mean = s * (1.0f / 512.0f);
        float var  = fmaxf((s2 - 512.0f * mean * mean) * (1.0f / 511.0f), 0.0f);
        float rden = 1.0f / (sqrtf(var) + EPSV);
        #pragma unroll
        for (int i = 0; i < 16; ++i) {
            int col = i * 32 + lane;
            float cn = gc[col] * ((v[i] - mean) * rden) + btc[col];
            float ot = gbuf[m * 2048 + 1024 + col];
            out_c[(size_t)(m0 + m) * HH + col] = cn;
            out_h[(size_t)(m0 + m) * HH + col] = ot * cn;
        }
    }
}

extern "C" void kernel_launch(void* const* d_in, const int* in_sizes, int n_in,
                              void* d_out, int out_size, void* d_ws, size_t ws_size,
                              hipStream_t stream) {
    const float* x   = (const float*)d_in[0];
    const float* ph  = (const float*)d_in[1];
    const float* pc  = (const float*)d_in[2];
    const float* Wf  = (const float*)d_in[3];
    const float* bf_ = (const float*)d_in[4];
    const float* Wi  = (const float*)d_in[5];
    const float* bi_ = (const float*)d_in[6];
    const float* Wo  = (const float*)d_in[7];
    const float* bo_ = (const float*)d_in[8];
    const float* Wj  = (const float*)d_in[9];
    const float* bj_ = (const float*)d_in[10];
    const float* gf  = (const float*)d_in[11];
    const float* btf = (const float*)d_in[12];
    const float* gi  = (const float*)d_in[13];
    const float* bti = (const float*)d_in[14];
    const float* go  = (const float*)d_in[15];
    const float* bto = (const float*)d_in[16];
    const float* gj  = (const float*)d_in[17];
    const float* btj = (const float*)d_in[18];
    const float* gc  = (const float*)d_in[19];
    const float* btc = (const float*)d_in[20];

    __bf16* wpack = (__bf16*)d_ws;             // 4 gates * 512*1024 bf16 = 4 MiB
    int rows = in_sizes[2] / HH;               // B (65536)
    float* out_h = (float*)d_out;
    float* out_c = out_h + (size_t)rows * HH;

    // Pack/convert weights into WMMA B-fragment layout (bf16).
    pack_weights_kernel<<<4096, 512, 0, stream>>>(Wf, Wi, Wo, Wj, wpack);

    // Fused GEMM + LayerNorm LSTM cell.
    size_t smem_bytes = 32768 + (size_t)16 * 2048 * sizeof(float); // 160 KiB
    lstm_kernel<<<rows / MTILE, 512, smem_bytes, stream>>>(
        x, ph, pc, wpack,
        bf_, bi_, bo_, bj_,
        gf, btf, gi, bti, go, bto, gj, btj, gc, btc,
        out_h, out_c);
}
